// SimpleRetention_66623532696062
// MI455X (gfx1250) — matrix operand order repaired and checked by reading, not compiled, not run on hardware
//
#include <hip/hip_runtime.h>

#define H   256
#define SEQ 8192
#define NB  4

typedef __attribute__((ext_vector_type(16))) __bf16         bf16x16;
typedef __attribute__((ext_vector_type(8)))  float          f32x8;
typedef __attribute__((ext_vector_type(8)))  unsigned short u16x8;
typedef __attribute__((ext_vector_type(16))) unsigned short u16x16;

// ---------- CDNA5 async copy (global -> LDS, ASYNCcnt-tracked) ----------

#if defined(__has_builtin)
#  if __has_builtin(__builtin_amdgcn_s_wait_asynccnt)
#    define WAIT_ASYNC() __builtin_amdgcn_s_wait_asynccnt(0)
#  endif
#endif
#ifndef WAIT_ASYNC
#  define WAIT_ASYNC() asm volatile("s_wait_asynccnt 0x0" ::: "memory")
#endif

// Per-lane 16B transfer: LDS[lds_byte_addr] = MEM[gaddr]  (GV mode, saddr=off)
__device__ __forceinline__ void async_copy_b128(unsigned lds_byte_addr, const void* gptr) {
  asm volatile("global_load_async_to_lds_b128 %0, %1, off"
               :: "v"(lds_byte_addr), "v"((unsigned long long)(uintptr_t)gptr)
               : "memory");
}

// ---------- helpers ----------

__device__ __forceinline__ unsigned short f2bf(float f) {
  unsigned int u = __float_as_uint(f);
  u += 0x7FFFu + ((u >> 16) & 1u);          // round-to-nearest-even
  return (unsigned short)(u >> 16);
}

__device__ __forceinline__ bf16x16 join16(u16x8 a, u16x8 b) {
  u16x16 r;
#pragma unroll
  for (int i = 0; i < 8; i++) { r[i] = a[i]; r[i + 8] = b[i]; }
  return __builtin_bit_cast(bf16x16, r);
}

// A-operand, 16x32 bf16 (ISA 7.12.2): lane holds row (l&15);
// lanes 0-15: K in {0..7,16..23}; lanes 16-31: K in {8..15,24..31}.
__device__ __forceinline__ bf16x16 load_matA(const unsigned short* rowp, int half) {
  u16x8 a = *(const u16x8*)(rowp + half * 8);
  u16x8 b = *(const u16x8*)(rowp + 16 + half * 8);
  return join16(a, b);
}

// B-operand, 32x16 bf16: lane holds col (l&15); lanes 0-15: K in {0..7,16..23};
// lanes 16-31: K in {8..15,24..31} (same order as A).
__device__ __forceinline__ bf16x16 load_matB(const unsigned short* colp, int half) {
  u16x8 a = *(const u16x8*)(colp + half * 8);
  u16x8 b = *(const u16x8*)(colp + 16 + half * 8);
  return join16(a, b);
}

__device__ __forceinline__ f32x8 wmma_bf16(bf16x16 a, bf16x16 b, f32x8 c) {
  return __builtin_amdgcn_wmma_f32_16x16x32_bf16(false, a, false, b, (short)0, c,
                                                 false, false);
}

// ---------- kernel 0: weights -> bf16, transposed (Wt[n][k] = W[k][n]) ----------

__global__ void wconv_kernel(const float* __restrict__ WQ, const float* __restrict__ WK,
                             const float* __restrict__ WV, unsigned short* __restrict__ Wt) {
  const int n = threadIdx.x;       // 0..255
  const int k = blockIdx.x;        // 0..255
  const int mat = blockIdx.y;      // 0..2
  const float* W = (mat == 0) ? WQ : ((mat == 1) ? WK : WV);
  Wt[(size_t)mat * H * H + (size_t)n * H + k] = f2bf(W[(size_t)k * H + n]);
}

// ---------- kernel 1: Q/K/V projections + RoPE, bf16 outputs ----------

__global__ void __launch_bounds__(128)
qkv_rope_kernel(const float* __restrict__ X, const unsigned short* __restrict__ Wt,
                unsigned short* __restrict__ Qb, unsigned short* __restrict__ Kb,
                unsigned short* __restrict__ Vb) {
  __shared__ __align__(16) unsigned short Xs[16][H];   // 8 KB bf16 strip of X

  const int tid = threadIdx.x;
  const int l = tid & 31, w = tid >> 5;
  const int half = l >> 4, ln = l & 15;
  const int row0 = blockIdx.x * 16;

  // stage 16x256 fp32 -> bf16 in LDS
  {
    const int rr = tid >> 3;            // 0..15
    const int cc = (tid & 7) * 32;      // 0..224
    const float* xp = X + (size_t)(row0 + rr) * H + cc;
#pragma unroll
    for (int c = 0; c < 4; c++) {
      u16x8 v;
#pragma unroll
      for (int j = 0; j < 8; j++) v[j] = f2bf(xp[c * 8 + j]);
      *(u16x8*)&Xs[rr][cc + c * 8] = v;
    }
  }
  __syncthreads();

  // 48 tiles = {Q,K,V} x 16 col-tiles; 12 per wave
  for (int t = w; t < 48; t += 4) {
    const int mat = t >> 4;
    const int ct = t & 15;
    const int n = ct * 16 + ln;
    const unsigned short* wcol = Wt + (size_t)mat * H * H + (size_t)n * H;

    f32x8 acc = {0.f, 0.f, 0.f, 0.f, 0.f, 0.f, 0.f, 0.f};
#pragma unroll
    for (int ks = 0; ks < 8; ks++) {
      bf16x16 a = load_matA(&Xs[ln][ks * 32], half);
      bf16x16 b = load_matB(wcol + ks * 32, half);
      acc = wmma_bf16(a, b, acc);
    }

    unsigned short* dst = (mat == 0) ? Qb : ((mat == 1) ? Kb : Vb);
    const int mbase = half * 8;
    if (mat == 2) {
#pragma unroll
      for (int v = 0; v < 8; v++) {
        const int r = row0 + mbase + v;
        dst[(size_t)r * H + n] = f2bf(acc[v]);
      }
    } else {
      // torchtune-style interleaved-pair RoPE; pair partner sits in lane l^1
      const float invf = __expf(-(float)(n & ~1) * (9.210340371976184f / 256.0f));
#pragma unroll
      for (int v = 0; v < 8; v++) {
        const int r = row0 + mbase + v;
        const int s = r & (SEQ - 1);   // sequence position
        float sn, cs;
        __sincosf((float)s * invf, &sn, &cs);
        const float x = acc[v];
        const float p = __shfl_xor(x, 1, 32);
        const float y = (n & 1) ? (x * cs + p * sn) : (x * cs - p * sn);
        dst[(size_t)r * H + n] = f2bf(y);
      }
    }
  }
}

// ---------- kernel 2: flash-style retention ----------

#define LN_GAMMA (-0.031748698314580176f)   // ln(0.96875)

__global__ void __launch_bounds__(64)
retention_kernel(const unsigned short* __restrict__ Qb, const unsigned short* __restrict__ Kb,
                 const unsigned short* __restrict__ Vb, float* __restrict__ out) {
  __shared__ __align__(16) unsigned short Qs[32][H];      // 16 KB
  __shared__ __align__(16) unsigned short Ks[32][H];      // 16 KB
  __shared__ __align__(16) unsigned short Vt[H][32];      // 16 KB (V transposed)
  __shared__ __align__(16) unsigned short Ps[2][16][32];  // 2 KB per-wave scores

  const int tid = threadIdx.x;
  const int l = tid & 31, w = tid >> 5;
  const int half = l >> 4, ln = l & 15;
  const int q0 = blockIdx.x * 32;
  const int b = blockIdx.y;
  const size_t rowbase = (size_t)b * SEQ;

  // stage Q tile (32x256 bf16) once: async DMA straight into LDS
  {
    const unsigned short* src = Qb + (rowbase + q0) * H;
    const unsigned ldsq = (unsigned)(uintptr_t)&Qs[0][0];
#pragma unroll
    for (int c = 0; c < 16; c++) {
      const int off = (tid * 16 + c) * 8;         // element offset
      async_copy_b128(ldsq + off * 2, src + off); // 16B per lane
    }
  }

  f32x8 o[16];
#pragma unroll
  for (int i = 0; i < 16; i++) o[i] = (f32x8){0.f, 0.f, 0.f, 0.f, 0.f, 0.f, 0.f, 0.f};

  const int nkt = q0 / 32 + 1;
  for (int kt = 0; kt < nkt; kt++) {
    const int kbase = kt * 32;
    __syncthreads();   // prior tile fully consumed before overwriting LDS

    // stage K tile (32x256): async DMA to LDS (ASYNCcnt-tracked)
    {
      const unsigned short* src = Kb + (rowbase + kbase) * H;
      const unsigned ldsk = (unsigned)(uintptr_t)&Ks[0][0];
#pragma unroll
      for (int c = 0; c < 16; c++) {
        const int off = (tid * 16 + c) * 8;
        async_copy_b128(ldsk + off * 2, src + off);
      }
    }
    // stage V tile transposed: Vt[n][k] so B-operand reads are contiguous
    {
      const int k = tid >> 1;            // 0..31
      const int n0 = (tid & 1) * 128;    // 0 or 128
      const unsigned short* src = Vb + (rowbase + kbase + k) * H + n0;
#pragma unroll
      for (int c = 0; c < 16; c++) {
        u16x8 v = *(const u16x8*)(src + c * 8);
#pragma unroll
        for (int j = 0; j < 8; j++) Vt[n0 + c * 8 + j][k] = v[j];
      }
    }
    if (kt + 1 < nkt) {   // pull next tiles toward L2/L0 (global_prefetch_b8)
      __builtin_prefetch(Kb + (rowbase + kbase + 32) * H, 0, 1);
      __builtin_prefetch(Vb + (rowbase + kbase + 32) * H, 0, 1);
    }
    WAIT_ASYNC();      // this wave's async LDS writes landed (covers Q on kt==0)
    __syncthreads();   // all waves' staging visible

    // P = (Q_strip K_tile^T) * gamma^(q-k) * causal  -> Ps[w] is 16x32 bf16
    const int qrow = q0 + w * 16;
#pragma unroll
    for (int sub = 0; sub < 2; sub++) {
      f32x8 acc = {0.f, 0.f, 0.f, 0.f, 0.f, 0.f, 0.f, 0.f};
#pragma unroll
      for (int ks = 0; ks < 8; ks++) {
        bf16x16 a = load_matA(&Qs[w * 16 + ln][ks * 32], half);
        bf16x16 bb = load_matB(&Ks[sub * 16 + ln][ks * 32], half);
        acc = wmma_bf16(a, bb, acc);
      }
      const int kk = kbase + sub * 16 + ln;     // key index (C-layout: N = l&15)
#pragma unroll
      for (int v = 0; v < 8; v++) {
        const int m = half * 8 + v;             // C-layout: M = vgpr + 8*(l>=16)
        const int d = (qrow + m) - kk;
        const float f = (d >= 0) ? __expf((float)d * LN_GAMMA) : 0.0f;
        Ps[w][m][sub * 16 + ln] = f2bf(acc[v] * f);
      }
    }
    __syncthreads();   // C-layout -> A-layout round trip through LDS

    // O += P @ V_tile  (K-dim = 32 keys)
    bf16x16 pa = load_matA(&Ps[w][ln][0], half);
#pragma unroll
    for (int ct = 0; ct < 16; ct++) {
      bf16x16 vb = load_matB(&Vt[ct * 16 + ln][0], half);
      o[ct] = wmma_bf16(pa, vb, o[ct]);
    }
    __syncthreads();   // done with Ks/Vt before next tile overwrites them
  }

  // write fp32 output
#pragma unroll
  for (int ct = 0; ct < 16; ct++) {
#pragma unroll
    for (int v = 0; v < 8; v++) {
      const int m = half * 8 + v;
      out[(rowbase + q0 + w * 16 + m) * H + ct * 16 + ln] = o[ct][v];
    }
  }
}

// ---------- launch ----------

extern "C" void kernel_launch(void* const* d_in, const int* in_sizes, int n_in,
                              void* d_out, int out_size, void* d_ws, size_t ws_size,
                              hipStream_t stream) {
  (void)in_sizes; (void)n_in; (void)out_size; (void)ws_size;

  const float* X  = (const float*)d_in[0];
  const float* WQ = (const float*)d_in[1];
  const float* WK = (const float*)d_in[2];
  const float* WV = (const float*)d_in[3];
  float* out = (float*)d_out;

  // workspace layout (bf16/u16 elements): Wt(3*256*256) | Qb | Kb | Vb (each B*S*H)
  unsigned short* ws = (unsigned short*)d_ws;
  unsigned short* Wt = ws;
  unsigned short* Qb = ws + (size_t)3 * H * H;
  unsigned short* Kb = Qb + (size_t)NB * SEQ * H;
  unsigned short* Vb = Kb + (size_t)NB * SEQ * H;

  wconv_kernel<<<dim3(H, 3), H, 0, stream>>>(WQ, WK, WV, Wt);
  qkv_rope_kernel<<<dim3((NB * SEQ) / 16), 128, 0, stream>>>(X, Wt, Qb, Kb, Vb);
  retention_kernel<<<dim3(SEQ / 32, NB), 64, 0, stream>>>(Qb, Kb, Vb, out);
}
